// TokenLevelCompressor_37271726194770
// MI455X (gfx1250) — matrix operand order repaired and checked by reading, not compiled, not run on hardware
//
#include <hip/hip_runtime.h>
#include <hip/hip_bf16.h>
#include <stdint.h>

typedef __attribute__((ext_vector_type(16))) __bf16 bf16x16;
typedef __attribute__((ext_vector_type(4)))  __bf16 bf16x4;
typedef __attribute__((ext_vector_type(8)))  float  f32x8;
typedef __attribute__((ext_vector_type(4)))  float  f32x4;
typedef __attribute__((ext_vector_type(4)))  unsigned int u32x4;

#define TOKENS  32768   // B*S
#define SEQ     8192    // S (tokens per batch; blocks never cross this)
#define DMODEL  2048    // D
#define HDIM    128     // head dim
#define MTILE   64      // output tokens per workgroup
#define KCHUNK  64      // K staged in LDS per iteration
#define LDSROWS 68      // MTILE + 4 (extra prev-block rows for b-branch)
#define LDSK    72      // KCHUNK + 8 pad (stride 144B = 36 banks, conflict-free b128)
#define NBLOCKS (TOKENS / 4)   // 8192 output blocks total

union FragU { bf16x16 v; u32x4 u[2]; };

// ---------------------------------------------------------------------------
// Pre-pack the 4 weight matrices (f32 [D, HD]) into bf16 in this GEMM's per-lane
// WMMA B-fragment order (contiguous K, same as the A loads) (B = 32x16 bf16 per fragment):
//   lane l (l<16):  col n = l,     K = kc*32 + 0..15   (2 per dword, low first)
//   lane l (l>=16): col n = l-16,  K = kc*32 + 16..31
// Stored as wp[ntg][kc][lane][16] so the GEMM does two coalesced b128 loads.
// ntg = side*16 + kind*8 + ntile   (side: 0=a,1=b ; kind: 0=kv(c), 1=z)
// ---------------------------------------------------------------------------
__global__ void pack_weights_kernel(const float* __restrict__ wkva,
                                    const float* __restrict__ wkvb,
                                    const float* __restrict__ wza,
                                    const float* __restrict__ wzb,
                                    __bf16* __restrict__ wp) {
  int idx  = blockIdx.x * blockDim.x + threadIdx.x;   // 32*64*32 = 65536 lane-chunks
  int ntg  = idx >> 11;          // 0..31
  int kcw  = (idx >> 5) & 63;    // K/32 chunk
  int l    = idx & 31;           // lane
  int side = (ntg >> 4) & 1;
  int kind = (ntg >> 3) & 1;
  const float* w = side ? (kind ? wzb : wkvb) : (kind ? wza : wkva);
  int n  = ((ntg & 7) << 4) + (l & 15);
  int kb = (kcw << 5) + ((l >> 4) << 4);
  __bf16* dst = wp + (size_t)idx * 16;
#pragma unroll
  for (int j = 0; j < 16; ++j)
    dst[j] = (__bf16)w[(size_t)(kb + j) * HDIM + n];
}

// ---------------------------------------------------------------------------
// Fused: 4-way projection GEMM (bf16 WMMA, f32 accum) + windowed softmax
// compression.  One workgroup = 64 tokens = 16 output blocks.
// waves 0-3: a-branch rows [t0, t0+64)    -> c_a | z_a
// waves 4-7: b-branch rows [t0-4, t0+60)  -> c_b | z_b
// wave (w&3)=q owns feature columns [32q, 32q+32) for BOTH kinds (c and z).
// ---------------------------------------------------------------------------
__global__ __launch_bounds__(256, 1)
void fused_compress_kernel(const float* __restrict__ h,
                           const __bf16* __restrict__ wp,
                           const float* __restrict__ ba,
                           const float* __restrict__ bb,
                           float* __restrict__ out) {
  __shared__ __bf16 tileA[LDSROWS][LDSK];     // 9792 B  (bf16 h tile)
  __shared__ float  eph[2][2][32][68];        // 34816 B (transposed epilogue dump)

  const int tid  = threadIdx.x;
  const int lane = tid & 31;
  const int wid  = tid >> 5;
  const int side = wid >> 2;            // 0 = a-branch, 1 = b-branch
  const int q    = wid & 3;             // 32-wide feature-column chunk
  const int l15  = lane & 15;
  const int lhi  = lane >> 4;
  const int aoff = side ? 0 : 4;        // a-branch rows start 4 deep in the tile
  const int t0   = blockIdx.x * MTILE;

  // n-tile ids this wave computes: t=0,1 -> kind kv (c), t=2,3 -> kind z
  int ntg[4];
#pragma unroll
  for (int t = 0; t < 4; ++t)
    ntg[t] = side * 16 + ((t >> 1) << 3) + 2 * q + (t & 1);

  f32x8 acc[4][4];                      // [m-tile][n-tile] : 128 VGPRs
#pragma unroll
  for (int mt = 0; mt < 4; ++mt)
#pragma unroll
    for (int t = 0; t < 4; ++t)
      acc[mt][t] = (f32x8)0.0f;

  // ---------------- K loop: stage h tile (f32 -> bf16) then WMMA -------------
  for (int kc0 = 0; kc0 < DMODEL; kc0 += KCHUNK) {
    for (int idx = tid; idx < LDSROWS * 16; idx += 256) {
      int row = idx >> 4, k4 = idx & 15;
      long gr = (long)t0 - 4 + row;
      if (gr < 0) gr = 0;               // OOB rows only feed the masked block 0
      const float* src = h + gr * (long)DMODEL + kc0 + k4 * 4;
      f32x4 v = *(const f32x4*)src;
      if (kc0 + KCHUNK < DMODEL)
        __builtin_prefetch(src + KCHUNK, 0, 0);   // global_prefetch_b8, speculative
      bf16x4 p;
      p[0] = (__bf16)v[0]; p[1] = (__bf16)v[1];
      p[2] = (__bf16)v[2]; p[3] = (__bf16)v[3];
      *(bf16x4*)&tileA[row][k4 * 4] = p;
    }
    __syncthreads();

#pragma unroll
    for (int ks = 0; ks < KCHUNK; ks += 32) {
      const int kcw = (kc0 + ks) >> 5;
      // B fragments: coalesced b128 pairs from pre-swizzled L2-hot weights
      bf16x16 bfrag[4];
#pragma unroll
      for (int t = 0; t < 4; ++t) {
        const __bf16* bp = wp + (((size_t)ntg[t] * 64 + kcw) * 32 + lane) * 16;
        FragU tmp;
        tmp.u[0] = *(const u32x4*)(bp);
        tmp.u[1] = *(const u32x4*)(bp + 8);
        bfrag[t] = tmp.v;
      }
      // A fragments from LDS (contiguous K = 16*lhi + 0..15, matching packed B), then 16 WMMAs
#pragma unroll
      for (int mt = 0; mt < 4; ++mt) {
        int row = aoff + mt * 16 + l15;
        int kb  = ks + lhi * 16;
        FragU af;
        af.u[0] = *(const u32x4*)&tileA[row][kb];
        af.u[1] = *(const u32x4*)&tileA[row][kb + 8];
#pragma unroll
        for (int t = 0; t < 4; ++t)
          acc[mt][t] = __builtin_amdgcn_wmma_f32_16x16x32_bf16(
              false, af.v, false, bfrag[t], (short)0, acc[mt][t], false, false);
      }
    }
    __syncthreads();
  }

  // ---------------- Fused epilogue: windowed softmax compression -------------
  // 4 passes over 32-feature chunks; pass p: waves with q==p dump transposed
  // tiles, then all 256 threads compute 16 blocks x 32 features.
  const bool firstblk = (t0 & (SEQ - 1)) == 0;   // block 0 of a sequence

  for (int p = 0; p < 4; ++p) {
    if (q == p) {
#pragma unroll
      for (int t = 0; t < 4; ++t) {
        int kind = t >> 1;
        int col  = ((t & 1) << 4) + l15;         // local col within 32-chunk
#pragma unroll
        for (int mt = 0; mt < 4; ++mt)
#pragma unroll
          for (int v = 0; v < 8; ++v)            // C layout: M = v + 8*lhi
            eph[side][kind][col][mt * 16 + v + lhi * 8] = acc[mt][t][v];
      }
    }
    __syncthreads();

#pragma unroll
    for (int r = 0; r < 2; ++r) {
      int p2 = tid + (r << 8);        // 512 (block, feature) pairs this pass
      int i  = p2 >> 5;               // block 0..15
      int cl = p2 & 31;               // local feature
      int f  = (p << 5) + cl;         // global feature 0..127
      f32x4 ca = *(const f32x4*)&eph[0][0][cl][i * 4];
      f32x4 za = *(const f32x4*)&eph[0][1][cl][i * 4];
      f32x4 cb = *(const f32x4*)&eph[1][0][cl][i * 4];
      f32x4 zb = *(const f32x4*)&eph[1][1][cl][i * 4];

      float la[4], lb[4];
#pragma unroll
      for (int j = 0; j < 4; ++j) {
        la[j] = za[j] + ba[j * HDIM + f];
        lb[j] = zb[j] + bb[j * HDIM + f];
      }
      const bool bvalid = !(firstblk && i == 0);
      float mx = la[0];
#pragma unroll
      for (int j = 1; j < 4; ++j) mx = fmaxf(mx, la[j]);
      if (bvalid)
#pragma unroll
        for (int j = 0; j < 4; ++j) mx = fmaxf(mx, lb[j]);
      float den = 0.f, num = 0.f;
#pragma unroll
      for (int j = 0; j < 4; ++j) {
        float e = __expf(la[j] - mx);
        den += e; num += e * ca[j];
      }
      if (bvalid)
#pragma unroll
        for (int j = 0; j < 4; ++j) {
          float e = __expf(lb[j] - mx);
          den += e; num += e * cb[j];
        }
      float o = num / den;
      size_t oi = (size_t)(t0 / 4 + i) * HDIM + f;
      out[oi] = o;                                   // compressed
      out[oi + (size_t)NBLOCKS * HDIM] = o;          // tuple duplicate
    }
    __syncthreads();
  }
}

extern "C" void kernel_launch(void* const* d_in, const int* in_sizes, int n_in,
                              void* d_out, int out_size, void* d_ws, size_t ws_size,
                              hipStream_t stream) {
  const float* h    = (const float*)d_in[0];
  const float* wkva = (const float*)d_in[1];
  const float* wkvb = (const float*)d_in[2];
  const float* wza  = (const float*)d_in[3];
  const float* wzb  = (const float*)d_in[4];
  const float* ba   = (const float*)d_in[5];
  const float* bb   = (const float*)d_in[6];
  float* out = (float*)d_out;
  __bf16* wp = (__bf16*)d_ws;   // 2 MB packed bf16 weights

  pack_weights_kernel<<<256, 256, 0, stream>>>(wkva, wkvb, wza, wzb, wp);
  fused_compress_kernel<<<TOKENS / MTILE, 256, 0, stream>>>(h, wp, ba, bb, out);
}
